// EfficientCrossAttention1D_16750372454957
// MI455X (gfx1250) — matrix operand order repaired and checked by reading, not compiled, not run on hardware
//
#include <hip/hip_runtime.h>

// ---------------------------------------------------------------------------
// EfficientCrossAttention1D on MI455X (gfx1250, wave32, WMMA bf16)
// B=8, C=512, L=8192, H=8, DK=DV=64
// All GEMMs in "TN" form: both operands K-contiguous per row, fragments load
// straight from global with b128s -> no LDS transpose, no barriers in GEMMs.
// ---------------------------------------------------------------------------

typedef unsigned short u16;
typedef __attribute__((ext_vector_type(8)))  unsigned short u16x8;   // 16 B
typedef __attribute__((ext_vector_type(4)))  float          f32x4;   // 16 B
typedef __attribute__((ext_vector_type(16))) __bf16         bf16x16; // 32 B
typedef __attribute__((ext_vector_type(8)))  float          f32x8;   // 32 B

union FragU { u16x8 h[2]; bf16x16 bf; };

__device__ __forceinline__ u16 f2bf(float f) {
    unsigned u = __builtin_bit_cast(unsigned, f);
    unsigned r = u + 0x7FFFu + ((u >> 16) & 1u);   // round-to-nearest-even
    return (u16)(r >> 16);
}
__device__ __forceinline__ float bf2f(u16 h) {
    unsigned u = ((unsigned)h) << 16;
    return __builtin_bit_cast(float, u);
}

// ---------------------------------------------------------------------------
// 1a) plain fp32 -> bf16 (weights)
// ---------------------------------------------------------------------------
__global__ void f32_to_bf16_kernel(const float* __restrict__ src,
                                   u16* __restrict__ dst, long long n) {
    long long i = (long long)blockIdx.x * blockDim.x + threadIdx.x;
    long long stride = (long long)gridDim.x * blockDim.x;
    for (; i < n; i += stride) dst[i] = f2bf(src[i]);
}

// ---------------------------------------------------------------------------
// 1b) transposing fp32 [C,L] -> bf16 [L,C] per batch (64x64 tiles via LDS)
// ---------------------------------------------------------------------------
__global__ __launch_bounds__(256)
void transpose_conv_kernel(const float* __restrict__ src, u16* __restrict__ dst,
                           int Cc, int Ln) {
    __shared__ u16 tile[64 * 72];                 // [c][l], padded rows
    const long long z = blockIdx.z;
    const int l0 = blockIdx.x * 64, c0 = blockIdx.y * 64;
    const float* s = src + z * (long long)Cc * Ln;
    u16* d = dst + z * (long long)Cc * Ln;
    const int t = threadIdx.x;
    const int r = t >> 2;            // 0..63
    const int q = (t & 3) * 16;      // 16-elem segment

    // load 16 floats of row (c0+r), cols l0+q.. ; convert; contiguous LDS store
    u16x8 lo, hi;
#pragma unroll
    for (int i = 0; i < 8; ++i) {
        lo[i] = f2bf(s[(long long)(c0 + r) * Ln + l0 + q + i]);
        hi[i] = f2bf(s[(long long)(c0 + r) * Ln + l0 + q + 8 + i]);
    }
    *(u16x8*)&tile[r * 72 + q]     = lo;
    *(u16x8*)&tile[r * 72 + q + 8] = hi;
    __syncthreads();

    // gather a column strip and store 16 contiguous c per thread
    const int lr = t >> 2;           // l row 0..63
    const int cs = (t & 3) * 16;     // c segment
    u16x8 a, b;
#pragma unroll
    for (int i = 0; i < 8; ++i) {
        a[i] = tile[(cs + i)     * 72 + lr];
        b[i] = tile[(cs + 8 + i) * 72 + lr];
    }
    u16* dp = &d[(long long)(l0 + lr) * Cc + c0 + cs];
    *(u16x8*)dp       = a;
    *(u16x8*)(dp + 8) = b;
}

// ---------------------------------------------------------------------------
// 2) TN WMMA GEMM: Y[z][m][n] = sum_k A[z][m][k] * Bt[z][n][k]
//    256 threads = 8 wave32s in 2x4; tile M=64 x N=128; K-step 32.
//    Fragments loaded directly from global (b128); no LDS, no barriers.
//    Per-z base offsets: off = (z/Hq)*s1 + (z%Hq)*s2  (covers (b,h) planes).
// ---------------------------------------------------------------------------
#define MT 64
#define NT 128
#define KT 32

__global__ __launch_bounds__(256)
void gemm_tn_kernel(const u16* __restrict__ A, int lda, long long sA1, long long sA2,
                    const u16* __restrict__ Bt, int ldb, long long sB1, long long sB2,
                    int Hq,
                    const float* __restrict__ bias, int biasCol,
                    const float* __restrict__ resid,
                    void* __restrict__ outp, int outBf16, int ldo,
                    long long sO1, long long sO2,
                    int Kd, int Nn) {
    const int t    = threadIdx.x;
    const int lane = t & 31;
    const int wave = t >> 5;
    const int wm   = wave >> 2;            // 0..1
    const int wn   = wave & 3;             // 0..3
    const int m0   = blockIdx.x * MT;
    const int n0   = blockIdx.y * NT;
    const long long z  = blockIdx.z;
    const long long zq = z / Hq, zr = z - zq * Hq;

    const u16* Ag = A  + zq * sA1 + zr * sA2;
    const u16* Bg = Bt + zq * sB1 + zr * sB2;
    const long long offO = zq * sO1 + zr * sO2;

    const int koffA = (lane >> 4) * 8;     // {0,8}
    const int koffB = (lane >> 4) * 8;     // {0,8}
    const int rsel  = lane & 15;

    // per-fragment row pointers (clamp B rows when N < tile width)
    const u16* ap[2];
    const u16* bp[2];
#pragma unroll
    for (int mi = 0; mi < 2; ++mi)
        ap[mi] = Ag + (long long)(m0 + wm * 32 + mi * 16 + rsel) * lda;
#pragma unroll
    for (int ni = 0; ni < 2; ++ni) {
        int rowb = n0 + wn * 32 + ni * 16 + rsel;
        if (rowb >= Nn) rowb = 0;          // clamped read; write is guarded
        bp[ni] = Bg + (long long)rowb * ldb;
    }

    f32x8 acc[2][2];
#pragma unroll
    for (int i = 0; i < 2; ++i)
#pragma unroll
        for (int j = 0; j < 2; ++j)
            acc[i][j] = (f32x8){0.f,0.f,0.f,0.f,0.f,0.f,0.f,0.f};

    for (int kk = 0; kk < Kd; kk += KT) {
        FragU a[2], b[2];
#pragma unroll
        for (int mi = 0; mi < 2; ++mi) {
            a[mi].h[0] = *(const u16x8*)(ap[mi] + kk + koffA);
            a[mi].h[1] = *(const u16x8*)(ap[mi] + kk + 16 + koffA);
        }
#pragma unroll
        for (int ni = 0; ni < 2; ++ni) {
            b[ni].h[0] = *(const u16x8*)(bp[ni] + kk + koffB);
            b[ni].h[1] = *(const u16x8*)(bp[ni] + kk + 16 + koffB);
        }
        if (kk + KT < Kd) {                // warm L0/L2 for next K-slab
            __builtin_prefetch(ap[0] + kk + KT + koffA, 0, 0);
            __builtin_prefetch(bp[0] + kk + KT + koffB, 0, 0);
        }
#pragma unroll
        for (int mi = 0; mi < 2; ++mi)
#pragma unroll
            for (int ni = 0; ni < 2; ++ni)
                acc[mi][ni] = __builtin_amdgcn_wmma_f32_16x16x32_bf16(
                    false, a[mi].bf, false, b[ni].bf,
                    (short)0, acc[mi][ni], false, false);
    }

    // writeback: D layout: M = r + 8*(lane>=16), N = lane&15
    const int cn  = lane & 15;
    const int mhi = (lane >> 4) * 8;
#pragma unroll
    for (int mi = 0; mi < 2; ++mi) {
#pragma unroll
        for (int ni = 0; ni < 2; ++ni) {
            const int col     = n0 + wn * 32 + ni * 16 + cn;
            const int rowBase = m0 + wm * 32 + mi * 16 + mhi;
            if (col < Nn) {
#pragma unroll
                for (int r = 0; r < 8; ++r) {
                    const int row = rowBase + r;
                    const long long idx = offO + (long long)row * ldo + col;
                    float v = acc[mi][ni][r];
                    if (bias) v += bias[biasCol ? col : row];
                    if (outBf16) {
                        ((u16*)outp)[idx] = f2bf(v);
                    } else {
                        if (resid) v += resid[idx];
                        ((float*)outp)[idx] = v;
                    }
                }
            }
        }
    }
}

// ---------------------------------------------------------------------------
// 3) Softmax over L: one block per (b,c) row of 8192 bf16 (in place on [C,L])
// ---------------------------------------------------------------------------
__global__ __launch_bounds__(256)
void softmax_rows_kernel(u16* __restrict__ data, int Ln) {
    __shared__ float red[256];
    const int t = threadIdx.x;
    u16* row = data + (long long)blockIdx.x * Ln;
    const int base = t * 32;

    float v[32];
#pragma unroll
    for (int i = 0; i < 4; ++i) {
        u16x8 h = *(const u16x8*)&row[base + i * 8];
#pragma unroll
        for (int j = 0; j < 8; ++j) v[i * 8 + j] = bf2f(h[j]);
    }
    float m = -3.4e38f;
#pragma unroll
    for (int i = 0; i < 32; ++i) m = fmaxf(m, v[i]);
    red[t] = m; __syncthreads();
    for (int s = 128; s > 0; s >>= 1) { if (t < s) red[t] = fmaxf(red[t], red[t + s]); __syncthreads(); }
    m = red[0]; __syncthreads();

    float sum = 0.f;
#pragma unroll
    for (int i = 0; i < 32; ++i) { v[i] = __expf(v[i] - m); sum += v[i]; }
    red[t] = sum; __syncthreads();
    for (int s = 128; s > 0; s >>= 1) { if (t < s) red[t] += red[t + s]; __syncthreads(); }
    const float inv = 1.f / red[0];

#pragma unroll
    for (int i = 0; i < 4; ++i) {
        u16x8 h;
#pragma unroll
        for (int j = 0; j < 8; ++j) h[j] = f2bf(v[i * 8 + j] * inv);
        *(u16x8*)&row[base + i * 8] = h;
    }
}

// ---------------------------------------------------------------------------
// 4) Softmax over head channels on QpT [B,L,C]: per (l,h) the 64 dk values
//    are CONTIGUOUS -> pure register softmax, no LDS.
// ---------------------------------------------------------------------------
__global__ __launch_bounds__(256)
void softmax_qT_kernel(u16* __restrict__ QpT, int Cc, int Ln) {
    const int t = threadIdx.x;
    const long long bz = blockIdx.x;              // batch
    const int l = blockIdx.y * 32 + (t >> 3);     // 32 l-rows per block
    const int h = t & 7;
    u16* p = QpT + (bz * Ln + l) * (long long)Cc + h * 64;

    u16x8 hv[8];
#pragma unroll
    for (int i = 0; i < 8; ++i) hv[i] = *(const u16x8*)(p + i * 8);
    float v[64];
    float m = -3.4e38f;
#pragma unroll
    for (int i = 0; i < 8; ++i)
#pragma unroll
        for (int j = 0; j < 8; ++j) { v[i * 8 + j] = bf2f(hv[i][j]); m = fmaxf(m, v[i * 8 + j]); }
    float sum = 0.f;
#pragma unroll
    for (int i = 0; i < 64; ++i) { v[i] = __expf(v[i] - m); sum += v[i]; }
    const float inv = 1.f / sum;
#pragma unroll
    for (int i = 0; i < 8; ++i) {
        u16x8 o;
#pragma unroll
        for (int j = 0; j < 8; ++j) o[j] = f2bf(v[i * 8 + j] * inv);
        *(u16x8*)(p + i * 8) = o;
    }
}

// ---------------------------------------------------------------------------
// 5) ctx = softmax(K) @ V^T per (b,h): M=64 x N=64, K=8192; 8 waves split K,
//    direct b128 global fragments, LDS ds_add_f32 reduction, emit ctx^T bf16.
// ---------------------------------------------------------------------------
__global__ __launch_bounds__(256)
void ctx_gemm_kernel(const u16* __restrict__ Ksm, const u16* __restrict__ V,
                     u16* __restrict__ ctxT, int Ln) {
    __shared__ float red[64 * 64];
    const int t = threadIdx.x, lane = t & 31, wave = t >> 5;
    const long long z = blockIdx.x;               // b*H + h
    const u16* Kb = Ksm + z * 64 * (long long)Ln;
    const u16* Vb = V   + z * 64 * (long long)Ln;

    f32x8 acc[4][4];
#pragma unroll
    for (int i = 0; i < 4; ++i)
#pragma unroll
        for (int j = 0; j < 4; ++j)
            acc[i][j] = (f32x8){0.f,0.f,0.f,0.f,0.f,0.f,0.f,0.f};

    const int kChunk = Ln >> 3;
    const int koffA = (lane >> 4) * 8;
    const int koffB = (lane >> 4) * 8;
    const int rsel  = lane & 15;

    for (int kk = wave * kChunk; kk < (wave + 1) * kChunk; kk += 32) {
        FragU a[4], b[4];
#pragma unroll
        for (int mi = 0; mi < 4; ++mi) {
            const u16* p = &Kb[(long long)(mi * 16 + rsel) * Ln + kk];
            a[mi].h[0] = *(const u16x8*)(p + koffA);
            a[mi].h[1] = *(const u16x8*)(p + 16 + koffA);
        }
#pragma unroll
        for (int ni = 0; ni < 4; ++ni) {
            const u16* p = &Vb[(long long)(ni * 16 + rsel) * Ln + kk];
            b[ni].h[0] = *(const u16x8*)(p + koffB);
            b[ni].h[1] = *(const u16x8*)(p + 16 + koffB);
        }
#pragma unroll
        for (int mi = 0; mi < 4; ++mi)
#pragma unroll
            for (int ni = 0; ni < 4; ++ni)
                acc[mi][ni] = __builtin_amdgcn_wmma_f32_16x16x32_bf16(
                    false, a[mi].bf, false, b[ni].bf,
                    (short)0, acc[mi][ni], false, false);
    }

    for (int i = t; i < 4096; i += 256) red[i] = 0.f;
    __syncthreads();
    const int mhi = (lane >> 4) * 8;
#pragma unroll
    for (int mi = 0; mi < 4; ++mi)
#pragma unroll
        for (int ni = 0; ni < 4; ++ni)
#pragma unroll
            for (int r = 0; r < 8; ++r)
                atomicAdd(&red[(mi * 16 + mhi + r) * 64 + ni * 16 + (lane & 15)],
                          acc[mi][ni][r]);
    __syncthreads();
    for (int i = t; i < 4096; i += 256) {         // ctxT[v][k] = ctx[k][v]
        const int v = i >> 6, k = i & 63;
        ctxT[z * 4096 + i] = f2bf(red[k * 64 + v]);
    }
}

// ---------------------------------------------------------------------------
// Host orchestration
// ---------------------------------------------------------------------------
extern "C" void kernel_launch(void* const* d_in, const int* in_sizes, int n_in,
                              void* d_out, int out_size, void* d_ws, size_t ws_size,
                              hipStream_t stream) {
    (void)in_sizes; (void)n_in; (void)out_size; (void)ws_size;
    const int Bc = 8, Cc = 512, Lc = 8192, Hc = 8;
    const long long nBCL = (long long)Bc * Cc * Lc;
    const long long nW   = (long long)Cc * Cc;
    const long long sCL  = (long long)Cc * Lc;        // per-batch plane

    const float* input_   = (const float*)d_in[0];
    const float* context_ = (const float*)d_in[1];
    const float* Wk = (const float*)d_in[2]; const float* bk = (const float*)d_in[3];
    const float* Wq = (const float*)d_in[4]; const float* bq = (const float*)d_in[5];
    const float* Wv = (const float*)d_in[6]; const float* bv = (const float*)d_in[7];
    const float* Wr = (const float*)d_in[8]; const float* br = (const float*)d_in[9];

    char* p = (char*)d_ws;
    auto carve = [&](long long bytes) {
        void* r = (void*)p; p += (bytes + 255) & ~255LL; return r;
    };
    u16* XcT  = (u16*)carve(nBCL * 2);   // context^T bf16  [B,L,C]
    u16* XiT  = (u16*)carve(nBCL * 2);   // input^T  bf16  [B,L,C]
    u16* Wkb  = (u16*)carve(nW * 2);
    u16* Wqb  = (u16*)carve(nW * 2);
    u16* Wvb  = (u16*)carve(nW * 2);
    u16* Wrb  = (u16*)carve(nW * 2);
    u16* Kp   = (u16*)carve(nBCL * 2);   // K proj [B,C,L], softmaxed in place
    u16* Vp   = (u16*)carve(nBCL * 2);   // V proj [B,C,L]
    u16* QpT  = (u16*)carve(nBCL * 2);   // Q proj^T [B,L,C], softmaxed in place
    u16* AttT = (u16*)carve(nBCL * 2);   // attended^T [B,L,C]
    u16* CtxT = (u16*)carve((long long)Bc * Hc * 64 * 64 * 2);

    // 1) converts (weights plain; activations transposed)
    f32_to_bf16_kernel<<<256, 256, 0, stream>>>(Wk, Wkb, nW);
    f32_to_bf16_kernel<<<256, 256, 0, stream>>>(Wq, Wqb, nW);
    f32_to_bf16_kernel<<<256, 256, 0, stream>>>(Wv, Wvb, nW);
    f32_to_bf16_kernel<<<256, 256, 0, stream>>>(Wr, Wrb, nW);
    transpose_conv_kernel<<<dim3(Lc / 64, Cc / 64, Bc), 256, 0, stream>>>(context_, XcT, Cc, Lc);
    transpose_conv_kernel<<<dim3(Lc / 64, Cc / 64, Bc), 256, 0, stream>>>(input_,   XiT, Cc, Lc);

    // 2) projections
    //    K = Wk @ context + bk  -> Kp [C,L]
    gemm_tn_kernel<<<dim3(Cc / MT, Lc / NT, Bc), 256, 0, stream>>>(
        Wkb, Cc, 0, 0, XcT, Cc, sCL, 0, 1, bk, 0, nullptr,
        Kp, 1, Lc, sCL, 0, Cc, Lc);
    //    V = Wv @ context + bv  -> Vp [C,L]
    gemm_tn_kernel<<<dim3(Cc / MT, Lc / NT, Bc), 256, 0, stream>>>(
        Wvb, Cc, 0, 0, XcT, Cc, sCL, 0, 1, bv, 0, nullptr,
        Vp, 1, Lc, sCL, 0, Cc, Lc);
    //    Q^T = input^T @ Wq^T + bq  -> QpT [L,C] (bias per column)
    gemm_tn_kernel<<<dim3(Lc / MT, Cc / NT, Bc), 256, 0, stream>>>(
        XiT, Cc, sCL, 0, Wqb, Cc, 0, 0, 1, bq, 1, nullptr,
        QpT, 1, Cc, sCL, 0, Cc, Cc);

    // 3) softmaxes
    softmax_rows_kernel<<<Bc * Cc, 256, 0, stream>>>(Kp, Lc);
    softmax_qT_kernel<<<dim3(Bc, Lc / 32), 256, 0, stream>>>(QpT, Cc, Lc);

    // 4) ctx^T per (b,h)
    ctx_gemm_kernel<<<Bc * Hc, 256, 0, stream>>>(Kp, Vp, CtxT, Lc);

    // 5) att^T[l][v] = sum_k QpT[l][k] * ctxT[v][k]   (per (b,h) plane)
    //    z = b*H + h;  A/out offset = b*L*C + h*64;  Bt offset = z*4096
    gemm_tn_kernel<<<dim3(Lc / MT, 1, Bc * Hc), 256, 0, stream>>>(
        QpT, Cc, (long long)Lc * Cc, 64,
        CtxT, 64, (long long)Hc * 4096, 4096, Hc,
        nullptr, 0, nullptr,
        AttT, 1, Cc, (long long)Lc * Cc, 64, 64, 64);

    // 6) out = Wr @ att + br + input_   (fp32, [B,C,L])
    gemm_tn_kernel<<<dim3(Cc / MT, Lc / NT, Bc), 256, 0, stream>>>(
        Wrb, Cc, 0, 0, AttT, Cc, sCL, 0, 1, br, 0, input_,
        (float*)d_out, 0, Lc, sCL, 0, Cc, Lc);
}
